// Decoder_53480932770653
// MI455X (gfx1250) — matrix operand order repaired and checked by reading, not compiled, not run on hardware
//
#include <hip/hip_runtime.h>
#include <hip/hip_bf16.h>

typedef _Float16 f16;
typedef __attribute__((ext_vector_type(16))) _Float16 v16h;
typedef __attribute__((ext_vector_type(8)))  float    v8f;
typedef __attribute__((ext_vector_type(4)))  float    v4f;

union FragH  { v16h f; v4f q[2]; };
union Pack8H { f16 h[8]; v4f q;  };

// ---------------------------------------------------------------------------
// Generic WMMA GEMM:  Cf[col*ldc + m] = bias[m] + sum_taps sum_k A_tap[m][k] * Bt[srcRow(col,tap)][k]
//   A  : fp16 [ntaps][M][K] row-major (weights)
//   Bt : fp16 [rows][K]     row-major (activations, position-major)
//   zrow: K fp16 zeros — out-of-range columns read from here (branch-free K loop)
//   column->source-row map: b = col/Lout, t = col%Lout, s = t + tap*dil - pad,
//                           row = b*Lin + s if 0<=s<Lin else zero row.
// One wave computes a 16(M) x 64(N) tile: 4 accumulators reuse each A fragment.
// ---------------------------------------------------------------------------
__global__ __launch_bounds__(128) void wmma_gemm_kernel(
    const f16* __restrict__ A, const f16* __restrict__ Bt,
    const f16* __restrict__ zrow,
    const float* __restrict__ bias, float* __restrict__ Cf, f16* __restrict__ Ch,
    int M, int K, int ldc, int Lout, int Lin, int ntaps, int dil, int pad,
    int Ntotal)
{
  const int lane  = threadIdx.x & 31;
  const int wave  = threadIdx.x >> 5;
  const int mtile = blockIdx.y * 4 + wave;
  if (mtile * 16 >= M) return;               // wave-uniform exit: EXEC stays full
  const int m0    = mtile * 16;
  const int n0    = blockIdx.x * 64;
  const int lrow  = lane & 15;
  const int abase = (lane < 16) ? 0 : 8;     // A frag: K pairs {0..7,16..23} / {8..15,24..31}
  const int bbase = (lane < 16) ? 0 : 8;     // B frag: K {0..7,16..23} / {8..15,24..31}

  // Per-column (b, t); invalid columns get a hugely negative t so every
  // tap's range test fails and the zero row is used. Division done once.
  int bb[4], tt[4];
#pragma unroll
  for (int j = 0; j < 4; ++j) {
    int col = n0 + j * 16 + lrow;
    if (col < Ntotal) {
      bb[j] = col / Lout;
      tt[j] = col - bb[j] * Lout;
    } else {
      bb[j] = 0;
      tt[j] = INT_MIN / 4;
    }
  }

  v8f zero = {};
  v8f acc[4] = {zero, zero, zero, zero};
  const f16* zp = zrow + bbase;

  for (int tap = 0; tap < ntaps; ++tap) {
    const f16* arow = A + (size_t)tap * M * K + (size_t)(m0 + lrow) * K + abase;
    const f16* brow[4];
#pragma unroll
    for (int j = 0; j < 4; ++j) {
      int s = tt[j] + tap * dil - pad;
      brow[j] = (s >= 0 && s < Lin) ? (Bt + (size_t)(bb[j] * Lin + s) * K + bbase)
                                    : zp;
    }
#pragma unroll 4
    for (int k0 = 0; k0 < K; k0 += 32) {
      FragH a;
      a.q[0] = *(const v4f*)(arow + k0);
      a.q[1] = *(const v4f*)(arow + k0 + 16);
      __builtin_prefetch(arow + k0 + 256, 0, 0);   // global_prefetch_b8
#pragma unroll
      for (int j = 0; j < 4; ++j) {
        FragH b;
        b.q[0] = *(const v4f*)(brow[j] + k0);
        b.q[1] = *(const v4f*)(brow[j] + k0 + 16);
        acc[j] = __builtin_amdgcn_wmma_f32_16x16x32_f16(
            false, a.f, false, b.f, (short)0, acc[j], false, false);
      }
    }
  }

  // C fragment: lane<16 holds rows m0+0..7 of column n; lane>=16 rows m0+8..15.
  const int moff = m0 + ((lane < 16) ? 0 : 8);
  float bv[8];
#pragma unroll
  for (int r = 0; r < 8; ++r) bv[r] = bias[moff + r];
#pragma unroll
  for (int j = 0; j < 4; ++j) {
    int col = n0 + j * 16 + lrow;
    if (col < Ntotal) {
      float vals[8];
#pragma unroll
      for (int r = 0; r < 8; ++r) vals[r] = acc[j][r] + bv[r];
      float* op = Cf + (size_t)col * ldc + moff;
      v4f s0 = {vals[0], vals[1], vals[2], vals[3]};
      v4f s1 = {vals[4], vals[5], vals[6], vals[7]};
      *(v4f*)(op)     = s0;
      *(v4f*)(op + 4) = s1;
      if (Ch) {
        Pack8H u;
#pragma unroll
        for (int r = 0; r < 8; ++r) u.h[r] = (f16)vals[r];
        *(v4f*)(Ch + (size_t)col * ldc + moff) = u.q;
      }
    }
  }
}

// ------------------- small helper kernels -------------------

__global__ void zero_fill_f16(f16* __restrict__ out, int n) {
  int i = blockIdx.x * blockDim.x + threadIdx.x;
  if (i < n) out[i] = (f16)0.f;
}

// strided fp32 -> fp16 (weight conversion; stride=3 extracts dilated-conv taps)
__global__ void cvt_f32_f16_strided(const float* __restrict__ in, f16* __restrict__ out,
                                    int n, int stride, int off) {
  int i = blockIdx.x * blockDim.x + threadIdx.x;
  if (i < n) out[i] = (f16)in[(size_t)i * stride + off];
}

// h1 = relu(latent @ w1^T + b1)   [32 x 512], K=128
__global__ void ti1_kernel(const float* __restrict__ latent, const float* __restrict__ w1,
                           const float* __restrict__ b1, float* __restrict__ h1) {
  int i = blockIdx.x * blockDim.x + threadIdx.x;
  if (i >= 32 * 512) return;
  int b = i >> 9, c = i & 511;
  float s = b1[c];
  for (int k = 0; k < 128; ++k) s += latent[b * 128 + k] * w1[c * 128 + k];
  h1[i] = fmaxf(s, 0.f);
}

// h = h1 @ w2^T + b2   [32 x 512], K=512
__global__ void ti2_kernel(const float* __restrict__ h1, const float* __restrict__ w2,
                           const float* __restrict__ b2, float* __restrict__ h) {
  int i = blockIdx.x * blockDim.x + threadIdx.x;
  if (i >= 32 * 512) return;
  int b = i >> 9, c = i & 511;
  float s = b2[c];
  for (int k = 0; k < 512; ++k) s += h1[b * 512 + k] * w2[c * 512 + k];
  h[i] = s;
}

// X0 fp16 [b][pos 0..256][k 0..1023]; pos0=[h;h], pos t+1=[emb[helper];h]
__global__ void build_x0_kernel(const float* __restrict__ emb, const int* __restrict__ helper,
                                const float* __restrict__ h, f16* __restrict__ X0) {
  int idx = blockIdx.x * blockDim.x + threadIdx.x;
  if (idx >= 32 * 257 * 1024) return;
  int k = idx & 1023;
  int pos = idx >> 10;
  int b = pos / 257;
  int p = pos - b * 257;
  float v;
  if (p == 0)        v = h[b * 512 + (k & 511)];
  else if (k < 512)  v = emb[(size_t)helper[b * 256 + (p - 1)] * 512 + k];
  else               v = h[b * 512 + (k - 512)];
  X0[idx] = (f16)v;
}

// per-channel mean / invstd over positions (C fixed at 512)
__global__ void bn_reduce_kernel(const float* __restrict__ y, int npos,
                                 float* __restrict__ mean, float* __restrict__ invstd) {
  __shared__ float ss[256], ss2[256];
  int c = blockIdx.x;
  float s = 0.f, s2 = 0.f;
  for (int p = threadIdx.x; p < npos; p += 256) {
    float v = y[(size_t)p * 512 + c];
    s += v; s2 += v * v;
  }
  ss[threadIdx.x] = s; ss2[threadIdx.x] = s2;
  __syncthreads();
  for (int o = 128; o > 0; o >>= 1) {
    if (threadIdx.x < o) { ss[threadIdx.x] += ss[threadIdx.x + o]; ss2[threadIdx.x] += ss2[threadIdx.x + o]; }
    __syncthreads();
  }
  if (threadIdx.x == 0) {
    float m = ss[0] / (float)npos;
    float var = ss2[0] / (float)npos - m * m;
    mean[c] = m;
    invstd[c] = rsqrtf(var + 1e-5f);
  }
}

// out_h = relu(bn(y))  fp16, layout [pos][c]
__global__ void bn_apply_kernel(const float* __restrict__ y, const float* __restrict__ mean,
                                const float* __restrict__ invstd, const float* __restrict__ g,
                                const float* __restrict__ bb, f16* __restrict__ out, int total) {
  int i = blockIdx.x * blockDim.x + threadIdx.x;
  if (i >= total) return;
  int c = i & 511;
  float v = (y[i] - mean[c]) * invstd[c] * g[c] + bb[c];
  out[i] = (f16)fmaxf(v, 0.f);
}

// x += relu(bn(y))[:, :257]; refresh fp16 mirror
__global__ void bn_residual_kernel(const float* __restrict__ y, const float* __restrict__ mean,
                                   const float* __restrict__ invstd, const float* __restrict__ g,
                                   const float* __restrict__ bb, float* __restrict__ x,
                                   f16* __restrict__ xh, int Lout2) {
  int i = blockIdx.x * blockDim.x + threadIdx.x;
  if (i >= 32 * 257 * 512) return;
  int c = i & 511;
  int pos = i >> 9;
  int b = pos / 257;
  int t = pos - b * 257;
  float v = (y[((size_t)(b * Lout2 + t) << 9) + c] - mean[c]) * invstd[c] * g[c] + bb[c];
  float nx = x[i] + fmaxf(v, 0.f);
  x[i] = nx;
  xh[i] = (f16)nx;
}

// ------------------- host orchestration -------------------

extern "C" void kernel_launch(void* const* d_in, const int* in_sizes, int n_in,
                              void* d_out, int out_size, void* d_ws, size_t ws_size,
                              hipStream_t stream) {
  (void)in_sizes; (void)n_in; (void)out_size; (void)ws_size;
  const int Bn = 32, Tn = 256, Lp = 257, Cc = 512, Vv = 12000;
  static const int DIL[10] = {1, 2, 4, 8, 16, 1, 2, 4, 8, 16};

  const int*   helper = (const int*)  d_in[0];
  const float* latent = (const float*)d_in[1];
  const float* emb    = (const float*)d_in[2];
  const float* ti_w1  = (const float*)d_in[3];
  const float* ti_b1  = (const float*)d_in[4];
  const float* ti_w2  = (const float*)d_in[5];
  const float* ti_b2  = (const float*)d_in[6];
  const float* init_w = (const float*)d_in[7];
  const float* init_b = (const float*)d_in[8];
  const float* fc_w   = (const float*)d_in[9];
  const float* fc_b   = (const float*)d_in[10];

  // workspace carve (256B aligned)
  char* p = (char*)d_ws;
  auto alloc = [&](size_t bytes) -> char* {
    char* r = p;
    p += (bytes + 255) & ~(size_t)255;
    return r;
  };
  float* h1b   = (float*)alloc((size_t)Bn * Cc * 4);
  float* hb    = (float*)alloc((size_t)Bn * Cc * 4);
  f16*   zrow  = (f16*)  alloc((size_t)1024 * 2);
  f16*   Wfc   = (f16*)  alloc((size_t)Vv * Cc * 2);
  f16*   Winit = (f16*)  alloc((size_t)Cc * 1024 * 2);
  f16*   Wblk  = (f16*)  alloc((size_t)10 * 5 * Cc * Cc * 2); // [blk][c1,tap0,tap1,tap2,c3]
  f16*   X0    = (f16*)  alloc((size_t)Bn * Lp * 1024 * 2);
  float* xbuf  = (float*)alloc((size_t)Bn * Lp * Cc * 4);
  f16*   xh    = (f16*)  alloc((size_t)Bn * Lp * Cc * 2);
  float* ybuf  = (float*)alloc((size_t)Bn * (Lp + 32) * Cc * 4);
  f16*   rbuf  = (f16*)  alloc((size_t)Bn * (Lp + 32) * Cc * 2);
  float* meanb = (float*)alloc(Cc * 4);
  float* istdb = (float*)alloc(Cc * 4);

  auto cdiv = [](int a, int b) { return (a + b - 1) / b; };
  auto cvt = [&](const float* in, f16* out, int n, int stride, int off) {
    cvt_f32_f16_strided<<<cdiv(n, 256), 256, 0, stream>>>(in, out, n, stride, off);
  };
  auto gemm = [&](const f16* A, const f16* Bt, const float* bias, float* Cf, f16* Ch,
                  int M, int K, int ldc, int Lout, int Lin, int ntaps, int dil, int pad,
                  int Ntotal) {
    dim3 grid(cdiv(cdiv(Ntotal, 16), 4), cdiv(M / 16, 4));
    wmma_gemm_kernel<<<grid, 128, 0, stream>>>(A, Bt, zrow, bias, Cf, Ch, M, K, ldc,
                                               Lout, Lin, ntaps, dil, pad, Ntotal);
  };
  const size_t WSLOT = (size_t)Cc * Cc;

  // --- zero row + weight conversions (fp32 -> fp16) ---
  zero_fill_f16<<<4, 256, 0, stream>>>(zrow, 1024);
  cvt(fc_w, Wfc, Vv * Cc, 1, 0);
  cvt(init_w, Winit, Cc * 1024, 1, 0);
  for (int i = 0; i < 10; ++i) {
    int base = 11 + 12 * i;
    cvt((const float*)d_in[base + 0], Wblk + (i * 5 + 0) * WSLOT, Cc * Cc, 1, 0);
    for (int j = 0; j < 3; ++j)
      cvt((const float*)d_in[base + 4], Wblk + (i * 5 + 1 + j) * WSLOT, Cc * Cc, 3, j);
    cvt((const float*)d_in[base + 8], Wblk + (i * 5 + 4) * WSLOT, Cc * Cc, 1, 0);
  }

  // --- latent MLP + input assembly ---
  ti1_kernel<<<cdiv(Bn * Cc, 256), 256, 0, stream>>>(latent, ti_w1, ti_b1, h1b);
  ti2_kernel<<<cdiv(Bn * Cc, 256), 256, 0, stream>>>(h1b, ti_w2, ti_b2, hb);
  build_x0_kernel<<<cdiv(Bn * Lp * 1024, 256), 256, 0, stream>>>(emb, helper, hb, X0);

  // --- init 1x1 conv: x = W_init(512x1024) * X0 ---
  const int Npos = Bn * Lp; // 8224
  gemm(Winit, X0, init_b, xbuf, xh, Cc, 1024, Cc, Lp, Lp, 1, 0, 0, Npos);

  // --- residual blocks ---
  for (int i = 0; i < 10; ++i) {
    int base = 11 + 12 * i;
    int d = DIL[i], pad = 2 * d;
    int Lout2 = Lp + 2 * d;
    int N2 = Bn * Lout2;
    const f16* Wc1 = Wblk + (i * 5 + 0) * WSLOT;
    const f16* Wc2 = Wblk + (i * 5 + 1) * WSLOT; // 3 taps contiguous
    const f16* Wc3 = Wblk + (i * 5 + 4) * WSLOT;

    // c1 (1x1) -> bn1 -> relu
    gemm(Wc1, xh, (const float*)d_in[base + 1], ybuf, nullptr, Cc, Cc, Cc, Lp, Lp, 1, 0, 0, Npos);
    bn_reduce_kernel<<<Cc, 256, 0, stream>>>(ybuf, Npos, meanb, istdb);
    bn_apply_kernel<<<cdiv(Npos * Cc, 256), 256, 0, stream>>>(
        ybuf, meanb, istdb, (const float*)d_in[base + 2], (const float*)d_in[base + 3],
        rbuf, Npos * Cc);

    // c2 (3-tap dilated, padded) -> bn2 -> relu
    gemm(Wc2, rbuf, (const float*)d_in[base + 5], ybuf, nullptr, Cc, Cc, Cc, Lout2, Lp, 3, d, pad, N2);
    bn_reduce_kernel<<<Cc, 256, 0, stream>>>(ybuf, N2, meanb, istdb);
    bn_apply_kernel<<<cdiv(N2 * Cc, 256), 256, 0, stream>>>(
        ybuf, meanb, istdb, (const float*)d_in[base + 6], (const float*)d_in[base + 7],
        rbuf, N2 * Cc);

    // c3 (1x1 over grown length) -> bn3 -> relu -> trim -> residual add
    gemm(Wc3, rbuf, (const float*)d_in[base + 9], ybuf, nullptr, Cc, Cc, Cc, Lout2, Lout2, 1, 0, 0, N2);
    bn_reduce_kernel<<<Cc, 256, 0, stream>>>(ybuf, N2, meanb, istdb);
    bn_residual_kernel<<<cdiv(Npos * Cc, 256), 256, 0, stream>>>(
        ybuf, meanb, istdb, (const float*)d_in[base + 10], (const float*)d_in[base + 11],
        xbuf, xh, Lout2);
  }

  // --- final FC: logits[b][t][v] = fc_w * x[:, :, 1:] + fc_b  (shift via pad=-1) ---
  gemm(Wfc, xh, fc_b, (float*)d_out, nullptr, Vv, Cc, Vv, Tn, Lp, 1, 0, -1, Bn * Tn);
}
